// MambaSeqToSeq_21337397526646
// MI455X (gfx1250) — compile-verified, matrix operand order checked by reading, not run on hardware
//
#include <hip/hip_runtime.h>
#include <math.h>

// ---------------- problem constants (from reference) ----------------
static constexpr int dS  = 256;   // vocab / head dim
static constexpr int dH  = 768;   // hidden
static constexpr int dE  = 1536;  // expanded
static constexpr int dN  = 16;    // state dim
static constexpr int dK  = 4;     // conv kernel
static constexpr int dR  = 48;    // dt rank
static constexpr int dNL = 2;     // layers
static constexpr int dL  = 1024;  // sequence length
static constexpr int NCHUNK = 16;
static constexpr int CLEN   = dL / NCHUNK;
static constexpr int DBCW   = dR + 2 * dN;   // 80

typedef __attribute__((ext_vector_type(16))) _Float16 v16h;
typedef __attribute__((ext_vector_type(8)))  float    v8f;
typedef __attribute__((ext_vector_type(8)))  _Float16 h8;

enum { EPI_NONE = 0, EPI_BIAS = 1, EPI_BIAS_SOFTPLUS = 2 };

// ---------------- WMMA fragment load (ISA 7.12.2 16-bit 16x32 layout) ----
// lane 0..15  : row = base+lane,    halves 0..7 = K k0+0..7,  8..15 = K k0+16..23
// lane 16..31 : row = base+lane-16, halves 0..7 = K k0+8..15, 8..15 = K k0+24..31
template <bool KALIGNED>
__device__ __forceinline__ v16h load_frag(const _Float16* __restrict__ base,
                                          int ld, int row, int k0, int K) {
  const int lane = threadIdx.x & 31;
  const _Float16* p = base + (size_t)(row + (lane & 15)) * (size_t)ld;
  const int kk = k0 + ((lane >> 4) << 3);
  h8 lo = {}, hi = {};
  if (KALIGNED || kk < K)      lo = *(const h8*)(p + kk);
  if (KALIGNED || kk + 16 < K) hi = *(const h8*)(p + kk + 16);
  v16h f;
#pragma unroll
  for (int i = 0; i < 8; ++i) { f[i] = lo[i]; f[i + 8] = hi[i]; }
  return f;
}

__device__ __forceinline__ v8f wmma_f16(v16h a, v16h b, v8f c) {
  return __builtin_amdgcn_wmma_f32_16x16x32_f16(false, a, false, b, (short)0, c,
                                                false, false);
}

__device__ __forceinline__ void store_tile(float* __restrict__ C,
                                           _Float16* __restrict__ Ch,
                                           const float* __restrict__ bias,
                                           int ldc, int m, int n, v8f acc, int epi) {
  const int lane  = threadIdx.x & 31;
  const int col   = n + (lane & 15);
  const int rbase = m + ((lane >> 4) << 3);
  const float b = bias ? bias[col] : 0.f;
#pragma unroll
  for (int i = 0; i < 8; ++i) {
    float v = acc[i] + b;
    if (epi == EPI_BIAS_SOFTPLUS) v = (v > 20.f) ? v : log1pf(__expf(v));
    const size_t off = (size_t)(rbase + i) * (size_t)ldc + col;
    if (C)  C[off]  = v;
    if (Ch) Ch[off] = (_Float16)v;
  }
}

// 32x(16*NT) tile body: 2 A frags x NT B frags, last K-step peeled so the
// steady-state loop is: unconditional loads -> 2*NT chained WMMAs -> swap.
template <bool KALIGNED, int NT>
__device__ __forceinline__ void gemm_body(const _Float16* __restrict__ A,
                                          const _Float16* __restrict__ B,
                                          float* __restrict__ C,
                                          _Float16* __restrict__ Ch,
                                          const float* __restrict__ bias,
                                          int K, int lda, int ldb, int ldc,
                                          int m0, int n0, int epi) {
  const int Kp = KALIGNED ? K : ((K + 31) & ~31);
  v8f acc[2][NT];
#pragma unroll
  for (int mi = 0; mi < 2; ++mi)
#pragma unroll
    for (int ni = 0; ni < NT; ++ni) acc[mi][ni] = v8f{};

  v16h a0 = load_frag<KALIGNED>(A, lda, m0,      0, K);
  v16h a1 = load_frag<KALIGNED>(A, lda, m0 + 16, 0, K);
  v16h b[NT];
#pragma unroll
  for (int ni = 0; ni < NT; ++ni)
    b[ni] = load_frag<KALIGNED>(B, ldb, n0 + 16 * ni, 0, K);

  for (int k0 = 32; k0 < Kp; k0 += 32) {
    // next slab (unconditional -> statically counted loadcnt, overlaps WMMAs)
    v16h na0 = load_frag<KALIGNED>(A, lda, m0,      k0, K);
    v16h na1 = load_frag<KALIGNED>(A, lda, m0 + 16, k0, K);
    v16h nb[NT];
#pragma unroll
    for (int ni = 0; ni < NT; ++ni)
      nb[ni] = load_frag<KALIGNED>(B, ldb, n0 + 16 * ni, k0, K);

#pragma unroll
    for (int ni = 0; ni < NT; ++ni) {
      acc[0][ni] = wmma_f16(a0, b[ni], acc[0][ni]);
      acc[1][ni] = wmma_f16(a1, b[ni], acc[1][ni]);
    }
    a0 = na0; a1 = na1;
#pragma unroll
    for (int ni = 0; ni < NT; ++ni) b[ni] = nb[ni];
  }
  // peeled last K-step
#pragma unroll
  for (int ni = 0; ni < NT; ++ni) {
    acc[0][ni] = wmma_f16(a0, b[ni], acc[0][ni]);
    acc[1][ni] = wmma_f16(a1, b[ni], acc[1][ni]);
  }

#pragma unroll
  for (int ni = 0; ni < NT; ++ni) {
    store_tile(C, Ch, bias, ldc, m0,      n0 + 16 * ni, acc[0][ni], epi);
    store_tile(C, Ch, bias, ldc, m0 + 16, n0 + 16 * ni, acc[1][ni], epi);
  }
}

// ---------------- generic GEMM: C[M,N] = A[M,K] @ B[N,K]^T (+bias,+softplus)
template <bool KALIGNED>
__global__ void wmma_gemm(const _Float16* __restrict__ A,
                          const _Float16* __restrict__ B,
                          float* __restrict__ C, _Float16* __restrict__ Ch,
                          const float* __restrict__ bias,
                          int M, int N, int K, int lda, int ldb, int ldc, int epi) {
  // force tile id into an SGPR: all tile control flow is scalar, EXEC stays
  // all-ones around every WMMA (ISA 7.12 requirement)
  const int wave  = __builtin_amdgcn_readfirstlane(
      blockIdx.x * (blockDim.x >> 5) + (threadIdx.x >> 5));
  const int ntn2  = (N + 31) >> 5;
  const int total = (M >> 5) * ntn2;
  if (wave >= total) return;                 // scalar branch
  const int m0 = (wave / ntn2) << 5;
  const int n0 = (wave % ntn2) << 5;

  if (n0 + 16 < N)                           // scalar branch (N=80 edge)
    gemm_body<KALIGNED, 2>(A, B, C, Ch, bias, K, lda, ldb, ldc, m0, n0, epi);
  else
    gemm_body<KALIGNED, 1>(A, B, C, Ch, bias, K, lda, ldb, ldc, m0, n0, epi);
}

// ---------------- f32 -> f16 convert (weights, x and final h; per-launch) -------------
__global__ void cvt_f32_f16(const float* __restrict__ s, _Float16* __restrict__ d, int n) {
  int i = blockIdx.x * blockDim.x + threadIdx.x;
  if (i < n) d[i] = (_Float16)s[i];
}

// ---------------- residual add + RMSNorm; emits f16 for in_proj GEMM --------
__global__ void add_rmsnorm(const float* __restrict__ h, float* __restrict__ res,
                            const float* __restrict__ w, _Float16* __restrict__ hn_h,
                            int first) {
  const int row = blockIdx.x;              // dL rows, dH cols, block = 256
  const float* hr = h   + (size_t)row * dH;
  float*       rr = res + (size_t)row * dH;
  __shared__ float red[8];
  __shared__ float sscale;
  float vals[3];
  float ss = 0.f;
#pragma unroll
  for (int j = 0; j < 3; ++j) {
    int c = threadIdx.x + j * 256;
    float v = first ? hr[c] : (hr[c] + rr[c]);
    rr[c] = v;
    vals[j] = v;
    ss += v * v;
  }
#pragma unroll
  for (int o = 16; o > 0; o >>= 1) ss += __shfl_down(ss, o, 32);
  if ((threadIdx.x & 31) == 0) red[threadIdx.x >> 5] = ss;
  __syncthreads();
  if (threadIdx.x == 0) {
    float t = 0.f;
#pragma unroll
    for (int i = 0; i < 8; ++i) t += red[i];
    sscale = rsqrtf(t / (float)dH + 1e-5f);
  }
  __syncthreads();
  const float sc = sscale;
#pragma unroll
  for (int j = 0; j < 3; ++j) {
    int c = threadIdx.x + j * 256;
    hn_h[(size_t)row * dH + c] = (_Float16)(vals[j] * sc * w[c]);
  }
}

// ---------------- causal conv1d (K=4) + SiLU; fp32 + f16 outputs ------------
__global__ void conv_silu(const float* __restrict__ xz, const float* __restrict__ cw,
                          const float* __restrict__ cb, float* __restrict__ u,
                          _Float16* __restrict__ uh) {
  int idx = blockIdx.x * blockDim.x + threadIdx.x;   // dL*dE
  if (idx >= dL * dE) return;
  int t = idx / dE, e = idx % dE;
  float acc = cb[e];
#pragma unroll
  for (int k = 0; k < dK; ++k) {
    int tt = t + k - (dK - 1);
    if (tt >= 0) acc += cw[e * dK + k] * xz[(size_t)tt * (2 * dE) + e];
  }
  float sv = acc / (1.f + __expf(-acc));
  u[idx]  = sv;
  uh[idx] = (_Float16)sv;
}

// ---------------- chunked selective scan -----------------------------------
// pass1: local scan per chunk (zero init) -> chunk-final state + chunk decay
__global__ void scan_pass1(const float* __restrict__ dt, const float* __restrict__ u,
                           const float* __restrict__ dbc, const float* __restrict__ alog,
                           float* __restrict__ SLoc, float* __restrict__ Pp) {
  int idx = blockIdx.x * blockDim.x + threadIdx.x;   // NCHUNK * dE
  if (idx >= NCHUNK * dE) return;
  int c = idx / dE, e = idx % dE;
  float A[dN], s[dN];
#pragma unroll
  for (int n = 0; n < dN; ++n) { A[n] = -__expf(alog[e * dN + n]); s[n] = 0.f; }
  float dtsum = 0.f;
  for (int t = c * CLEN; t < (c + 1) * CLEN; ++t) {
    float d  = dt[(size_t)t * dE + e];
    float uu = u [(size_t)t * dE + e];
    dtsum += d;
    const float4* B4 = (const float4*)(dbc + (size_t)t * DBCW + dR);
    float4 bb[4];
#pragma unroll
    for (int q = 0; q < 4; ++q) bb[q] = B4[q];      // 16B uniform-addr loads
    const float* Bp = (const float*)bb;
    float du = d * uu;
#pragma unroll
    for (int n = 0; n < dN; ++n)
      s[n] = s[n] * __expf(d * A[n]) + du * Bp[n];
  }
#pragma unroll
  for (int n = 0; n < dN; ++n) {
    SLoc[(size_t)idx * dN + n] = s[n];
    Pp  [(size_t)idx * dN + n] = __expf(A[n] * dtsum);  // prod of exp(dt*A)
  }
}

// fix-up: tiny sequential pass over 16 chunks per (e,n)
__global__ void scan_fix(const float* __restrict__ SLoc, const float* __restrict__ Pp,
                         float* __restrict__ Sinit) {
  int idx = blockIdx.x * blockDim.x + threadIdx.x;   // dE * dN
  if (idx >= dE * dN) return;
  float s = 0.f;
  for (int c = 0; c < NCHUNK; ++c) {
    size_t off = (size_t)c * dE * dN + idx;
    Sinit[off] = s;
    s = s * Pp[off] + SLoc[off];
  }
}

// pass2: replay with true init, fuse y += u*D and *silu(z); emit f16 for out_proj
__global__ void scan_pass2(const float* __restrict__ dt, const float* __restrict__ u,
                           const float* __restrict__ dbc, const float* __restrict__ alog,
                           const float* __restrict__ Dw, const float* __restrict__ xz,
                           const float* __restrict__ Sinit, _Float16* __restrict__ yg) {
  int idx = blockIdx.x * blockDim.x + threadIdx.x;   // NCHUNK * dE
  if (idx >= NCHUNK * dE) return;
  int c = idx / dE, e = idx % dE;
  float A[dN], s[dN];
#pragma unroll
  for (int n = 0; n < dN; ++n) {
    A[n] = -__expf(alog[e * dN + n]);
    s[n] = Sinit[(size_t)c * dE * dN + (size_t)e * dN + n];
  }
  const float de = Dw[e];
  for (int t = c * CLEN; t < (c + 1) * CLEN; ++t) {
    float d  = dt[(size_t)t * dE + e];
    float uu = u [(size_t)t * dE + e];
    const float4* B4 = (const float4*)(dbc + (size_t)t * DBCW + dR);
    float4 bc[8];
#pragma unroll
    for (int q = 0; q < 8; ++q) bc[q] = B4[q];      // B (16) then C (16)
    const float* Bp = (const float*)bc;
    const float* Cp = Bp + dN;
    float du = d * uu;
    float y = 0.f;
#pragma unroll
    for (int n = 0; n < dN; ++n) {
      s[n] = s[n] * __expf(d * A[n]) + du * Bp[n];
      y += s[n] * Cp[n];
    }
    y += uu * de;
    float z = xz[(size_t)t * (2 * dE) + dE + e];
    float g = z / (1.f + __expf(-z));
    yg[(size_t)t * dE + e] = (_Float16)(y * g);
  }
}

// ---------------- host orchestration ----------------------------------------
extern "C" void kernel_launch(void* const* d_in, const int* in_sizes, int n_in,
                              void* d_out, int out_size, void* d_ws, size_t ws_size,
                              hipStream_t stream) {
  const float* x      = (const float*)d_in[0];
  const float* emb_w  = (const float*)d_in[1];
  const float* emb_b  = (const float*)d_in[2];
  const float* norm_w = (const float*)d_in[3];
  const float* in_w   = (const float*)d_in[4];
  const float* conv_w = (const float*)d_in[5];
  const float* conv_b = (const float*)d_in[6];
  const float* xp_w   = (const float*)d_in[7];
  const float* dt_w   = (const float*)d_in[8];
  const float* dt_b   = (const float*)d_in[9];
  const float* A_log  = (const float*)d_in[10];
  const float* Dw     = (const float*)d_in[11];
  const float* out_w  = (const float*)d_in[12];
  const float* head_w = (const float*)d_in[13];
  const float* head_b = (const float*)d_in[14];
  float* out = (float*)d_out;
  (void)in_sizes; (void)n_in; (void)out_size; (void)ws_size;

  // workspace carve-up (256B aligned regions)
  char* wp = (char*)d_ws;
  auto allocF = [&](size_t n) { float* p = (float*)wp;       wp += ((n * 4 + 255) & ~(size_t)255); return p; };
  auto allocH = [&](size_t n) { _Float16* p = (_Float16*)wp; wp += ((n * 2 + 255) & ~(size_t)255); return p; };

  float*    h     = allocF((size_t)dL * dH);
  float*    res   = allocF((size_t)dL * dH);
  float*    xz    = allocF((size_t)dL * 2 * dE);
  float*    u     = allocF((size_t)dL * dE);
  float*    dbc   = allocF((size_t)dL * DBCW);
  float*    dtb   = allocF((size_t)dL * dE);
  float*    SLoc  = allocF((size_t)NCHUNK * dE * dN);
  float*    Pp    = allocF((size_t)NCHUNK * dE * dN);
  float*    Sini  = allocF((size_t)NCHUNK * dE * dN);
  _Float16* hn_h  = allocH((size_t)dL * dH);
  _Float16* u_h   = allocH((size_t)dL * dE);
  _Float16* dbc_h = allocH((size_t)dL * DBCW);
  _Float16* yg_h  = allocH((size_t)dL * dE);
  _Float16* a_h   = allocH((size_t)dL * dH);        // x / final-h as GEMM A
  _Float16* w_h   = allocH((size_t)2 * dE * dH);    // largest weight (in_proj)

  auto cvt = [&](const float* s, _Float16* d, int n) {
    cvt_f32_f16<<<(n + 255) / 256, 256, 0, stream>>>(s, d, n);
  };
  auto gemm = [&](const _Float16* A, const _Float16* B, float* C, _Float16* Ch,
                  const float* bias, int M, int N, int K,
                  int lda, int ldb, int ldc, int epi) {
    int tiles  = (M / 32) * ((N + 31) / 32);
    int blocks = (tiles + 3) / 4;                 // 4 waves (128 threads) / block
    if (K % 32 == 0)
      wmma_gemm<true ><<<blocks, 128, 0, stream>>>(A, B, C, Ch, bias, M, N, K, lda, ldb, ldc, epi);
    else
      wmma_gemm<false><<<blocks, 128, 0, stream>>>(A, B, C, Ch, bias, M, N, K, lda, ldb, ldc, epi);
  };

  // ---- embedding: h = x @ emb_w.T + emb_b ----
  cvt(x, a_h, dL * dS);
  cvt(emb_w, w_h, dH * dS);
  gemm(a_h, w_h, h, nullptr, emb_b, dL, dH, dS, dS, dS, dH, EPI_BIAS);

  for (int i = 0; i < dNL; ++i) {
    // res (+)= h ; hn = rmsnorm(res) -> f16
    add_rmsnorm<<<dL, 256, 0, stream>>>(h, res, norm_w + (size_t)i * dH, hn_h, i == 0);

    // xz = hn @ in_w.T  (1024 x 3072)
    cvt(in_w + (size_t)i * 2 * dE * dH, w_h, 2 * dE * dH);
    gemm(hn_h, w_h, xz, nullptr, nullptr, dL, 2 * dE, dH, dH, dH, 2 * dE, EPI_NONE);

    // u = silu(causal_conv(xin))
    conv_silu<<<(dL * dE + 255) / 256, 256, 0, stream>>>(
        xz, conv_w + (size_t)i * dE * dK, conv_b + (size_t)i * dE, u, u_h);

    // dbc = u @ xp_w.T  (1024 x 80)
    cvt(xp_w + (size_t)i * DBCW * dE, w_h, DBCW * dE);
    gemm(u_h, w_h, dbc, dbc_h, nullptr, dL, DBCW, dE, dE, dE, DBCW, EPI_NONE);

    // dt = softplus(dbc[:, :48] @ dt_w.T + dt_b)   (K=48 tail-padded)
    cvt(dt_w + (size_t)i * dE * dR, w_h, dE * dR);
    gemm(dbc_h, w_h, dtb, nullptr, dt_b + (size_t)i * dE,
         dL, dE, dR, DBCW, dR, dE, EPI_BIAS_SOFTPLUS);

    // chunked selective scan -> yg (f16, gated)
    const float* al = A_log + (size_t)i * dE * dN;
    scan_pass1<<<(NCHUNK * dE + 255) / 256, 256, 0, stream>>>(dtb, u, dbc, al, SLoc, Pp);
    scan_fix  <<<(dE * dN + 255) / 256, 256, 0, stream>>>(SLoc, Pp, Sini);
    scan_pass2<<<(NCHUNK * dE + 255) / 256, 256, 0, stream>>>(
        dtb, u, dbc, al, Dw + (size_t)i * dE, xz, Sini, yg_h);

    // h = yg @ out_w.T  (1024 x 768)
    cvt(out_w + (size_t)i * dH * dE, w_h, dH * dE);
    gemm(yg_h, w_h, h, nullptr, nullptr, dL, dH, dE, dE, dE, dH, EPI_NONE);
  }

  // ---- head: out = h @ head_w.T + head_b ----
  cvt(h, a_h, dL * dH);
  cvt(head_w, w_h, dS * dH);
  gemm(a_h, w_h, out, nullptr, head_b, dL, dS, dH, dH, dH, dS, EPI_BIAS);
}
